// GRU_79843442032752
// MI455X (gfx1250) — matrix operand order repaired and checked by reading, not compiled, not run on hardware
//
#include <hip/hip_runtime.h>
#include <math.h>
#include <stdint.h>

#define VOCAB  32000
#define DMODEL 256
#define NH     1024
#define BB     16
#define TT     512
#define MROWS  (BB * TT)   // 8192

typedef __bf16 bf16;
typedef bf16  bf16x16 __attribute__((ext_vector_type(16)));
typedef float f32x8   __attribute__((ext_vector_type(8)));

union FragU { bf16x16 v; uint4 q[2]; };

__device__ __forceinline__ f32x8 wmma_bf16(bf16x16 a, bf16x16 b, f32x8 c) {
  // D = A(16x32) * B(32x16) + C, f32 accumulate
  return __builtin_amdgcn_wmma_f32_16x16x32_bf16(false, a, false, b, (short)0, c,
                                                 false, false);
}

// ------------------------------------------------ async global->LDS (CDNA5)
#if defined(__HIP_DEVICE_COMPILE__) && defined(__gfx1250__)
#define ASYNC_LDS 1
#endif

__device__ __forceinline__ void copy16_to_lds(const void* g, void* l) {
#ifdef ASYNC_LDS
  // GLOBAL_LOAD_ASYNC_TO_LDS_B128: vdst = LDS byte address (low 32 bits of the
  // flat shared pointer per the LDS aperture mapping), vaddr = 64-bit global.
  unsigned lds            = (unsigned)(size_t)l;
  unsigned long long gad  = (unsigned long long)(size_t)g;
  asm volatile("global_load_async_to_lds_b128 %0, %1, off"
               :: "v"(lds), "v"(gad) : "memory");
#else
  *(uint4*)l = *(const uint4*)g;
#endif
}

__device__ __forceinline__ void async_wait_all() {
#ifdef ASYNC_LDS
  asm volatile("s_wait_asynccnt 0x0" ::: "memory");
#endif
}

// ---------------------------------------------------------------- conversions
__global__ void k_cvt(const float* __restrict__ x, bf16* __restrict__ y, int n) {
  int i = blockIdx.x * blockDim.x + threadIdx.x;
  if (i < n) y[i] = (bf16)x[i];
}

__global__ void k_embed(const int* __restrict__ xb, const float* __restrict__ C,
                        bf16* __restrict__ emb) {
  int row = blockIdx.x;           // b*T + t
  int d   = threadIdx.x;          // 256 threads == DMODEL
  int tok = xb[row];
  emb[(size_t)row * DMODEL + d] = (bf16)C[(size_t)tok * DMODEL + d];
}

// ---------------------------------------------------------------- WMMA GEMM
// D[M,N] = A[M,K](bf16) * W[N,K](bf16)^T + bias[N],  f32 out.
// Block tile 128x128, 8 waves (4 along M x 2 along N), wave tile 32x64.
// K staged through double-buffered LDS in chunks of 64 via async loads.
#define BM 128
#define BN 128
#define KC 64
#define LDS_STR 72   // padded row stride (elements) -> 144B rows
#define GEMM_LDS_BYTES (2 * 2 * BM * LDS_STR * (int)sizeof(bf16))  // 73728

__device__ __forceinline__ void stage128x64(const bf16* __restrict__ G,
                                            bf16* S, int K, int tid) {
  // G points at (row0, kc); copy 128 rows x 64 cols, 16B per thread per iter
  for (int it = 0; it < 4; ++it) {
    int li = (tid + it * 256) * 8;     // 0..8191
    int r  = li >> 6;
    int c  = li & 63;
    copy16_to_lds(G + (size_t)r * K + c, S + r * LDS_STR + c);
  }
}

__global__ __launch_bounds__(256) void k_gemm(const bf16* __restrict__ A,
                                              const bf16* __restrict__ W,
                                              const float* __restrict__ bias,
                                              float* __restrict__ D,
                                              int M, int N, int K) {
  extern __shared__ __attribute__((aligned(16))) char gsm[];
  const int nbuf = BM * LDS_STR;            // elements per buffer
  bf16* As = (bf16*)gsm;                    // [2][128*LDS_STR]
  bf16* Bs = As + 2 * nbuf;                 // [2][128*LDS_STR]

  const int tid  = threadIdx.x;
  const int lane = tid & 31;
  const int wave = tid >> 5;
  const int wm   = wave & 3;   // 0..3  -> M offset wm*32
  const int wn   = wave >> 2;  // 0..1  -> N offset wn*64
  const int m0   = blockIdx.y * BM;
  const int n0   = blockIdx.x * BN;

  const f32x8 vzero = {0.f, 0.f, 0.f, 0.f, 0.f, 0.f, 0.f, 0.f};
  f32x8 acc[2][4];
  for (int i = 0; i < 2; ++i)
    for (int j = 0; j < 4; ++j) acc[i][j] = vzero;

  const int mrow = lane & 15;
  const int hi   = lane >> 4;

  const bf16* Ag = A + (size_t)m0 * K;
  const bf16* Wg = W + (size_t)n0 * K;

  // prologue: stage chunk 0 into buffer 0
  stage128x64(Ag, As, K, tid);
  stage128x64(Wg, Bs, K, tid);
  async_wait_all();
  __syncthreads();

  int buf = 0;
  for (int kc = 0; kc < K; kc += KC) {
    // issue async staging of the next chunk into the other buffer
    if (kc + KC < K) {
      stage128x64(Ag + kc + KC, As + (buf ^ 1) * nbuf, K, tid);
      stage128x64(Wg + kc + KC, Bs + (buf ^ 1) * nbuf, K, tid);
    }
    const bf16* Ab = As + buf * nbuf;
    const bf16* Bb = Bs + buf * nbuf;

    for (int kk = 0; kk < KC; kk += 32) {
      FragU a[2], b[4];
      const int kb = kk + (hi << 3);          // A: K split 8 + 8 per lane half
      for (int mt = 0; mt < 2; ++mt) {
        const bf16* pa = Ab + (wm * 32 + mt * 16 + mrow) * LDS_STR + kb;
        a[mt].q[0] = *(const uint4*)pa;        // K = kb..kb+7
        a[mt].q[1] = *(const uint4*)(pa + 16); // K = kb+16..kb+23
      }
      const int kb2 = kk + (hi << 3);         // B: K split 8 + 8 per lane half
      for (int nt = 0; nt < 4; ++nt) {
        const bf16* pb = Bb + (wn * 64 + nt * 16 + (lane & 15)) * LDS_STR + kb2;
        b[nt].q[0] = *(const uint4*)pb;
        b[nt].q[1] = *(const uint4*)(pb + 16);
      }
      for (int nt = 0; nt < 4; ++nt) {
        acc[0][nt] = wmma_bf16(a[0].v, b[nt].v, acc[0][nt]);
        acc[1][nt] = wmma_bf16(a[1].v, b[nt].v, acc[1][nt]);
      }
    }
    async_wait_all();
    __syncthreads();
    buf ^= 1;
  }

  // epilogue: lane holds col n = lane&15, rows {0..7}+8*hi per tile
  const int rb = hi << 3;
  for (int mt = 0; mt < 2; ++mt)
    for (int nt = 0; nt < 4; ++nt) {
      int col   = n0 + wn * 64 + nt * 16 + (lane & 15);
      float bi  = bias[col];
      int row0  = m0 + wm * 32 + mt * 16 + rb;
      for (int r = 0; r < 8; ++r)
        D[(size_t)(row0 + r) * N + col] = acc[mt][nt][r] + bi;
    }
}

// ---------------------------------------------------------------- recurrence
// Single persistent workgroup (32 waves), state in LDS (bf16), loops T steps.
// Phase A: gates[16,2048] = sigmoid(Wx_t + h @ U^T)     (128 tiles / 4 per wave)
// Phase B: c = tanh(Wcx_t + (r*h) @ Uc^T); h update     (64 tiles / 2 per wave)
#define HS (NH + 16)   // padded LDS row stride

__global__ __launch_bounds__(1024) void k_gru(const float* __restrict__ Wx,
                                              const float* __restrict__ Wcx,
                                              const bf16* __restrict__ U,
                                              const bf16* __restrict__ Uc,
                                              const float* __restrict__ h0,
                                              bf16* __restrict__ h_all) {
  extern __shared__ __attribute__((aligned(16))) char smem[];
  bf16* h_s  = (bf16*)smem;         // [16][HS] hidden state
  bf16* z_s  = h_s + 16 * HS;       // [16][HS] z gate
  bf16* rh_s = z_s + 16 * HS;       // [16][HS] r*h

  const int tid  = threadIdx.x;
  const int lane = tid & 31;
  const int wave = tid >> 5;        // 0..31
  const int mrow = lane & 15;
  const int hi   = lane >> 4;
  const f32x8 vzero = {0.f, 0.f, 0.f, 0.f, 0.f, 0.f, 0.f, 0.f};

  for (int i = tid; i < BB * NH; i += 1024) {
    int m = i >> 10, n = i & (NH - 1);
    h_s[m * HS + n] = (bf16)h0[n];
  }
  __syncthreads();

  for (int t = 0; t < TT; ++t) {
    // ---- Phase A: gate GEMM (N = 2048) ----
    f32x8 acc[4];
    for (int j = 0; j < 4; ++j) acc[j] = vzero;
    for (int k = 0; k < NH; k += 32) {
      FragU a, b[4];
      const bf16* pa = &h_s[mrow * HS + k + (hi << 3)];
      a.q[0] = *(const uint4*)pa;
      a.q[1] = *(const uint4*)(pa + 16);
      for (int j = 0; j < 4; ++j) {
        int n = (wave * 4 + j) * 16 + (lane & 15);
        const bf16* pb = &U[(size_t)n * NH + k + (hi << 3)];
        b[j].q[0] = *(const uint4*)pb;
        b[j].q[1] = *(const uint4*)(pb + 16);
      }
      for (int j = 0; j < 4; ++j)
        acc[j] = wmma_bf16(a.v, b[j].v, acc[j]);
    }
    for (int j = 0; j < 4; ++j) {
      int n = (wave * 4 + j) * 16 + (lane & 15);
      for (int r = 0; r < 8; ++r) {
        int m     = (hi << 3) + r;                       // batch row
        float wx  = Wx[(size_t)(m * TT + t) * (2 * NH) + n];
        float g   = 1.0f / (1.0f + expf(-(acc[j][r] + wx)));
        if (n < NH) {
          z_s[m * HS + n] = (bf16)g;
        } else {
          int hc = n - NH;
          rh_s[m * HS + hc] = (bf16)(g * (float)h_s[m * HS + hc]);
        }
      }
    }
    __syncthreads();

    // ---- Phase B: candidate GEMM (N = 1024) + state update ----
    f32x8 acc2[2];
    acc2[0] = vzero; acc2[1] = vzero;
    for (int k = 0; k < NH; k += 32) {
      FragU a, b[2];
      const bf16* pa = &rh_s[mrow * HS + k + (hi << 3)];
      a.q[0] = *(const uint4*)pa;
      a.q[1] = *(const uint4*)(pa + 16);
      for (int j = 0; j < 2; ++j) {
        int n = (wave * 2 + j) * 16 + (lane & 15);
        const bf16* pb = &Uc[(size_t)n * NH + k + (hi << 3)];
        b[j].q[0] = *(const uint4*)pb;
        b[j].q[1] = *(const uint4*)(pb + 16);
      }
      for (int j = 0; j < 2; ++j)
        acc2[j] = wmma_bf16(a.v, b[j].v, acc2[j]);
    }
    for (int j = 0; j < 2; ++j) {
      int n = (wave * 2 + j) * 16 + (lane & 15);
      for (int r = 0; r < 8; ++r) {
        int m    = (hi << 3) + r;
        float c  = tanhf(acc2[j][r] + Wcx[(size_t)(m * TT + t) * NH + n]);
        float z  = (float)z_s[m * HS + n];
        float hp = (float)h_s[m * HS + n];
        float hn = (1.0f - z) * hp + z * c;
        h_s[m * HS + n] = (bf16)hn;                       // owned uniquely
        h_all[(size_t)(m * TT + t) * NH + n] = (bf16)hn;
      }
    }
    __syncthreads();
  }
}

// ---------------------------------------------------------------- loss
__global__ void k_loss_row(const float* __restrict__ logits,
                           const int* __restrict__ targets,
                           float* __restrict__ row_loss) {
  __shared__ float red[256];
  const int row = blockIdx.x;
  const int tid = threadIdx.x;
  const float* p = logits + (size_t)row * VOCAB;

  float mx = -INFINITY;
  for (int i = tid; i < VOCAB; i += 256) mx = fmaxf(mx, p[i]);
  red[tid] = mx; __syncthreads();
  for (int s = 128; s > 0; s >>= 1) {
    if (tid < s) red[tid] = fmaxf(red[tid], red[tid + s]);
    __syncthreads();
  }
  mx = red[0]; __syncthreads();

  float sum = 0.f;
  for (int i = tid; i < VOCAB; i += 256) sum += expf(p[i] - mx);
  red[tid] = sum; __syncthreads();
  for (int s = 128; s > 0; s >>= 1) {
    if (tid < s) red[tid] += red[tid + s];
    __syncthreads();
  }
  if (tid == 0) {
    float lse = mx + logf(red[0]);
    row_loss[row] = lse - p[targets[row]];
  }
}

__global__ void k_loss_reduce(const float* __restrict__ rl, float* __restrict__ out) {
  __shared__ float red[256];
  float s = 0.f;
  for (int i = threadIdx.x; i < MROWS; i += 256) s += rl[i];
  red[threadIdx.x] = s; __syncthreads();
  for (int st = 128; st > 0; st >>= 1) {
    if (threadIdx.x < st) red[threadIdx.x] += red[threadIdx.x + st];
    __syncthreads();
  }
  if (threadIdx.x == 0) *out = red[0] / (float)MROWS;
}

// ---------------------------------------------------------------- launch
extern "C" void kernel_launch(void* const* d_in, const int* in_sizes, int n_in,
                              void* d_out, int out_size, void* d_ws, size_t ws_size,
                              hipStream_t stream) {
  const int*   xb   = (const int*)d_in[0];
  const int*   tgt  = (const int*)d_in[1];
  const float* C    = (const float*)d_in[2];
  const float* W_w  = (const float*)d_in[3];
  const float* W_b  = (const float*)d_in[4];
  const float* U_w  = (const float*)d_in[5];
  const float* Wc_w = (const float*)d_in[6];
  const float* Wc_b = (const float*)d_in[7];
  const float* Uc_w = (const float*)d_in[8];
  const float* V_w  = (const float*)d_in[9];
  const float* V_b  = (const float*)d_in[10];
  const float* h0   = (const float*)d_in[11];

  char* ws = (char*)d_ws;
  size_t off = 0;
  auto alloc = [&](size_t bytes) -> void* {
    void* p = ws + off;
    off = (off + bytes + 255) & ~(size_t)255;
    return p;
  };
  bf16*  emb      = (bf16*)alloc((size_t)MROWS * DMODEL * 2);
  bf16*  Wbf      = (bf16*)alloc((size_t)2 * NH * DMODEL * 2);
  bf16*  Wcbf     = (bf16*)alloc((size_t)NH * DMODEL * 2);
  bf16*  Ubf      = (bf16*)alloc((size_t)2 * NH * NH * 2);
  bf16*  Ucbf     = (bf16*)alloc((size_t)NH * NH * 2);
  bf16*  Vbf      = (bf16*)alloc((size_t)VOCAB * NH * 2);
  float* Wx       = (float*)alloc((size_t)MROWS * 2 * NH * 4);
  float* Wcx      = (float*)alloc((size_t)MROWS * NH * 4);
  bf16*  h_all    = (bf16*)alloc((size_t)MROWS * NH * 2);
  float* row_loss = (float*)alloc((size_t)MROWS * 4);

  // weight conversions f32 -> bf16
  k_cvt<<<(2 * NH * DMODEL + 255) / 256, 256, 0, stream>>>(W_w, Wbf, 2 * NH * DMODEL);
  k_cvt<<<(NH * DMODEL + 255) / 256, 256, 0, stream>>>(Wc_w, Wcbf, NH * DMODEL);
  k_cvt<<<(2 * NH * NH + 255) / 256, 256, 0, stream>>>(U_w, Ubf, 2 * NH * NH);
  k_cvt<<<(NH * NH + 255) / 256, 256, 0, stream>>>(Uc_w, Ucbf, NH * NH);
  k_cvt<<<(VOCAB * NH + 255) / 256, 256, 0, stream>>>(V_w, Vbf, VOCAB * NH);

  // embedding gather (bf16)
  k_embed<<<MROWS, DMODEL, 0, stream>>>(xb, C, emb);

  // input projections: Wx [8192,2048], Wcx [8192,1024]
  k_gemm<<<dim3(2 * NH / BN, MROWS / BM), 256, GEMM_LDS_BYTES, stream>>>(
      emb, Wbf, W_b, Wx, MROWS, 2 * NH, DMODEL);
  k_gemm<<<dim3(NH / BN, MROWS / BM), 256, GEMM_LDS_BYTES, stream>>>(
      emb, Wcbf, Wc_b, Wcx, MROWS, NH, DMODEL);

  // sequential GRU recurrence: one persistent workgroup, 99840B dynamic LDS
  k_gru<<<1, 1024, 3 * 16 * HS * (int)sizeof(bf16), stream>>>(
      Wx, Wcx, Ubf, Ucbf, h0, h_all);

  // logits [8192,32000] straight into d_out
  float* logits = (float*)d_out;
  k_gemm<<<dim3(VOCAB / BN, MROWS / BM), 256, GEMM_LDS_BYTES, stream>>>(
      h_all, Vbf, V_b, logits, MROWS, VOCAB, NH);

  // deterministic cross-entropy
  k_loss_row<<<MROWS, 256, 0, stream>>>(logits, tgt, row_loss);
  k_loss_reduce<<<1, 256, 0, stream>>>(row_loss, logits + (size_t)MROWS * VOCAB);
}
